// TransitionUp_56916906606937
// MI455X (gfx1250) — matrix operand order repaired and checked by reading, not compiled, not run on hardware
//
#include <hip/hip_runtime.h>

typedef __attribute__((ext_vector_type(16))) _Float16 v16h;
typedef __attribute__((ext_vector_type(8)))  float    v8f;

#define COUT 128
#define LN_EPS 1e-5f

// ---------------------------------------------------------------------------
// Transpose + f32->f16 convert of W[K,128] into Wt[128,K] so the WMMA
// B-fragment becomes a single contiguous 32-byte load per lane.
// ---------------------------------------------------------------------------
__global__ __launch_bounds__(256)
void transpose_f16_kernel(const float* __restrict__ W,
                          _Float16* __restrict__ Wt,
                          int K)
{
    int i = blockIdx.x * blockDim.x + threadIdx.x;   // over K*128 elements
    if (i >= K * COUT) return;
    int k = i / COUT;
    int n = i - k * COUT;
    Wt[n * K + k] = (_Float16)W[i];
}

// ---------------------------------------------------------------------------
// Fused GEMM (X[rows,K] @ W[K,128] + bias) -> LayerNorm(128) -> ReLU -> out.
// One block = 16 output rows x 128 cols. 256 threads = 8 wave32; wave w owns
// the 16x16 tile at columns 16*w. K templated so the loop unrolls x4 around
// v_wmma_f32_16x16x32_f16 (f32 accumulate).
//   A frag: two contiguous 8-float runs  -> 4x global_load_b128 + cvt_pk
//   B frag: 16 contiguous halves of Wt   -> 1x v16h load (2x b128)
// ---------------------------------------------------------------------------
template <int K>
__global__ __launch_bounds__(256)
void gemm_ln_relu_kernel(const float* __restrict__ X,
                         const _Float16* __restrict__ Wt,   // [128, K] f16
                         const float* __restrict__ bias,
                         const float* __restrict__ gamma,
                         const float* __restrict__ beta,
                         float* __restrict__ out)
{
    __shared__ float smem[16 * 132];  // 16 rows x 128 cols, +4 pad

    const int tid    = threadIdx.x;
    const int lane   = tid & 31;
    const int wave   = tid >> 5;
    const int laneHi = (lane >> 4) & 1;   // 0 = lanes 0-15, 1 = lanes 16-31
    const int lane16 = lane & 15;
    const int row0   = blockIdx.x * 16;
    const int col0   = wave * 16;
    const int m      = row0 + lane16;     // A-matrix row held by this lane
    const int n      = col0 + lane16;     // B/D column held by this lane

    const float*    Arow = X + (size_t)m * K;
    const _Float16* Brow = Wt + (size_t)n * K;

    v8f acc = {};

#pragma unroll 4
    for (int k0 = 0; k0 < K; k0 += 32) {
        // ---- A fragment: lane holds K = k0+laneHi*16+{0..15} ----
        const float4* pA0 = (const float4*)(Arow + k0 + laneHi * 16);
        const float4* pA1 = (const float4*)(Arow + k0 + 8 + laneHi * 16);
        float4 f0 = pA0[0], f1 = pA0[1];
        float4 f2 = pA1[0], f3 = pA1[1];
        v16h a;
        a[0]  = (_Float16)f0.x; a[1]  = (_Float16)f0.y;
        a[2]  = (_Float16)f0.z; a[3]  = (_Float16)f0.w;
        a[4]  = (_Float16)f1.x; a[5]  = (_Float16)f1.y;
        a[6]  = (_Float16)f1.z; a[7]  = (_Float16)f1.w;
        a[8]  = (_Float16)f2.x; a[9]  = (_Float16)f2.y;
        a[10] = (_Float16)f2.z; a[11] = (_Float16)f2.w;
        a[12] = (_Float16)f3.x; a[13] = (_Float16)f3.y;
        a[14] = (_Float16)f3.z; a[15] = (_Float16)f3.w;

        // ---- B fragment: lane holds K = k0+laneHi*16+{0..15}, column n ----
        v16h b = *(const v16h*)(Brow + k0 + laneHi * 16);

        acc = __builtin_amdgcn_wmma_f32_16x16x32_f16(
            /*neg_a=*/false, a, /*neg_b=*/false, b,
            /*c_mod=*/(short)0, acc, /*reuse_a=*/false, /*reuse_b=*/false);
    }

    // D layout: lanes0-15 vgpr r -> (M=r, N=lane); lanes16-31 -> (M=r+8).
    const float bn = bias[n];
#pragma unroll
    for (int r = 0; r < 8; ++r) {
        int row = r + laneHi * 8;
        smem[row * 132 + n] = acc[r] + bn;
    }
    __syncthreads();

    // LayerNorm over the 128-wide row: 16 threads per row, 8 elems each.
    const int row   = tid >> 4;   // 0..15
    const int cbase = tid & 15;
    float s = 0.f, s2 = 0.f;
#pragma unroll
    for (int j = 0; j < 8; ++j) {
        float v = smem[row * 132 + cbase + 16 * j];
        s += v; s2 += v * v;
    }
    // reduce across the 16 lanes of this half-wave (masks stay within half)
#pragma unroll
    for (int msk = 8; msk >= 1; msk >>= 1) {
        s  += __shfl_xor(s,  msk, 32);
        s2 += __shfl_xor(s2, msk, 32);
    }
    const float mu   = s * (1.0f / 128.0f);
    const float var  = s2 * (1.0f / 128.0f) - mu * mu;
    const float rstd = rsqrtf(var + LN_EPS);
#pragma unroll
    for (int j = 0; j < 8; ++j) {
        int cc  = cbase + 16 * j;
        float v = smem[row * 132 + cc];
        float y = (v - mu) * rstd * gamma[cc] + beta[cc];
        out[(row0 + row) * COUT + cc] = fmaxf(y, 0.0f);
    }
}

// ---------------------------------------------------------------------------
// Brute-force 3-NN: 1 thread per fine point; coarse points staged through LDS
// in 1024-point chunks; next chunk prefetched with global_prefetch. Outputs
// normalized inverse-square-distance weights + indices.
// ---------------------------------------------------------------------------
#define KNN_CHUNK 1024

__global__ __launch_bounds__(256)
void knn3_kernel(const float* __restrict__ pos,
                 const float* __restrict__ pos_sub,
                 int Ns,
                 int* __restrict__ idx_out,
                 float* __restrict__ w_out)
{
    __shared__ float sp[KNN_CHUNK * 3];

    const int i  = blockIdx.x * blockDim.x + threadIdx.x;
    const float px = pos[i * 3 + 0];
    const float py = pos[i * 3 + 1];
    const float pz = pos[i * 3 + 2];

    float d0 = 3.4e38f, d1 = 3.4e38f, d2 = 3.4e38f;
    int   i0 = 0, i1 = 0, i2 = 0;

    for (int base = 0; base < Ns; base += KNN_CHUNK) {
        __syncthreads();
        for (int t = threadIdx.x; t < KNN_CHUNK * 3; t += blockDim.x)
            sp[t] = pos_sub[base * 3 + t];
        __syncthreads();
        if (base + KNN_CHUNK < Ns)
            __builtin_prefetch(&pos_sub[(base + KNN_CHUNK) * 3], 0, 1);

#pragma unroll 4
        for (int j = 0; j < KNN_CHUNK; ++j) {
            float dx = px - sp[j * 3 + 0];
            float dy = py - sp[j * 3 + 1];
            float dz = pz - sp[j * 3 + 2];
            float d  = dx * dx + dy * dy + dz * dz;
            int  gj  = base + j;
            if (d < d2) {
                if (d < d1) {
                    d2 = d1; i2 = i1;
                    if (d < d0) { d1 = d0; i1 = i0; d0 = d; i0 = gj; }
                    else        { d1 = d;  i1 = gj; }
                } else { d2 = d; i2 = gj; }
            }
        }
    }

    const float w0 = 1.0f / fmaxf(d0, 1e-16f);
    const float w1 = 1.0f / fmaxf(d1, 1e-16f);
    const float w2 = 1.0f / fmaxf(d2, 1e-16f);
    const float inv = 1.0f / (w0 + w1 + w2);
    idx_out[i * 3 + 0] = i0;
    idx_out[i * 3 + 1] = i1;
    idx_out[i * 3 + 2] = i2;
    w_out[i * 3 + 0] = w0 * inv;
    w_out[i * 3 + 1] = w1 * inv;
    w_out[i * 3 + 2] = w2 * inv;
}

// ---------------------------------------------------------------------------
// Interpolation residual: one wave per output row; lane l handles float4 at
// channel 4*l. out[i] += w0*h[i0] + w1*h[i1] + w2*h[i2].
// ---------------------------------------------------------------------------
__global__ __launch_bounds__(256)
void interp_add_kernel(const float* __restrict__ h,
                       const int* __restrict__ idx,
                       const float* __restrict__ w,
                       float* __restrict__ out)
{
    const int wave = threadIdx.x >> 5;
    const int lane = threadIdx.x & 31;
    const int row  = blockIdx.x * 8 + wave;
    const int c    = lane * 4;

    const int   j0 = idx[row * 3 + 0];
    const int   j1 = idx[row * 3 + 1];
    const int   j2 = idx[row * 3 + 2];
    const float w0 = w[row * 3 + 0];
    const float w1 = w[row * 3 + 1];
    const float w2 = w[row * 3 + 2];

    const float4 a0 = *(const float4*)&h[j0 * COUT + c];
    const float4 a1 = *(const float4*)&h[j1 * COUT + c];
    const float4 a2 = *(const float4*)&h[j2 * COUT + c];

    float4 o = *(float4*)&out[row * COUT + c];
    o.x += w0 * a0.x + w1 * a1.x + w2 * a2.x;
    o.y += w0 * a0.y + w1 * a1.y + w2 * a2.y;
    o.z += w0 * a0.z + w1 * a1.z + w2 * a2.z;
    o.w += w0 * a0.w + w1 * a1.w + w2 * a2.w;
    *(float4*)&out[row * COUT + c] = o;
}

// ---------------------------------------------------------------------------
extern "C" void kernel_launch(void* const* d_in, const int* in_sizes, int n_in,
                              void* d_out, int out_size, void* d_ws, size_t ws_size,
                              hipStream_t stream)
{
    const float* x        = (const float*)d_in[0];   // [N, 128]
    const float* x_sub    = (const float*)d_in[1];   // [Ns, 256]
    const float* pos      = (const float*)d_in[2];   // [N, 3]
    const float* pos_sub  = (const float*)d_in[3];   // [Ns, 3]
    const float* W_sub    = (const float*)d_in[4];   // [256, 128]
    const float* b_sub    = (const float*)d_in[5];
    const float* g_sub    = (const float*)d_in[6];
    const float* beta_sub = (const float*)d_in[7];
    const float* W        = (const float*)d_in[8];   // [128, 128]
    const float* b        = (const float*)d_in[9];
    const float* g        = (const float*)d_in[10];
    const float* beta     = (const float*)d_in[11];
    float* out            = (float*)d_out;           // [N, 128]

    const int N   = in_sizes[0] / COUT;   // 32768
    const int Ns  = in_sizes[1] / 256;    // 8192
    const int Cin = 256;

    // workspace layout (all offsets 256B-aligned)
    char* wsp = (char*)d_ws;
    float* h      = (float*)wsp;                       // Ns*128 f32 = 4 MB
    wsp += (size_t)Ns * COUT * sizeof(float);
    int* idx3     = (int*)wsp;                         // N*3 i32
    wsp += (size_t)N * 3 * sizeof(int);
    float* w3     = (float*)wsp;                       // N*3 f32
    wsp += (size_t)N * 3 * sizeof(float);
    _Float16* Wt_sub = (_Float16*)wsp;                 // [128, 256] f16
    wsp += (size_t)COUT * Cin * sizeof(_Float16);
    _Float16* Wt     = (_Float16*)wsp;                 // [128, 128] f16

    // 0) pre-transpose/convert weights to f16 [n][k]
    transpose_f16_kernel<<<(Cin * COUT + 255) / 256, 256, 0, stream>>>(W_sub, Wt_sub, Cin);
    transpose_f16_kernel<<<(COUT * COUT + 255) / 256, 256, 0, stream>>>(W, Wt, COUT);

    // 1) h = relu(LN(x_sub @ W_sub + b_sub))
    gemm_ln_relu_kernel<256><<<Ns / 16, 256, 0, stream>>>(
        x_sub, Wt_sub, b_sub, g_sub, beta_sub, h);

    // 2) out = relu(LN(x @ W + b))
    gemm_ln_relu_kernel<128><<<N / 16, 256, 0, stream>>>(
        x, Wt, b, g, beta, out);

    // 3) 3-NN + inverse-distance weights
    knn3_kernel<<<N / 256, 256, 0, stream>>>(pos, pos_sub, Ns, idx3, w3);

    // 4) out += interp(h)
    interp_add_kernel<<<N / 8, 256, 0, stream>>>(h, idx3, w3, out);
}
